// SSTFRLayer_18124761989230
// MI455X (gfx1250) — compile-verified, matrix operand order checked by reading, not run on hardware
//
#include <hip/hip_runtime.h>
#include <hip/hip_bf16.h>
#include <stdint.h>

// ---- problem constants (reference: K = 1152, fixed by its constants) ----
#define DCH 128      // channels
#define KK  1152     // filter length
#define KB  36       // KK / 32 k-blocks
#define LT  80000    // time samples
#define BB  8        // batch
#define NT  64       // time columns per workgroup (4 sub-tiles of 16)
#define RB  16       // row blocks: 256 rows (128 real + 128 imag) / 16
#define SP  81152    // padded f16 x stride per batch: >= K-1+L, mult. of 64

typedef _Float16 v16h __attribute__((ext_vector_type(16)));
typedef float    v8f  __attribute__((ext_vector_type(8)));

static __device__ inline unsigned short f2h_bits(float f) {
    union { _Float16 h; unsigned short u; } cv;
    cv.h = (_Float16)f;                     // v_cvt_f16_f32, RNE
    return cv.u;
}

// -------------------------------------------------------------------------
// Kernel 1: filter bank in WMMA A-fragment-major layout (f16).
//   lane<16 : row M=lane,    j=0..7 -> K=kb*32+j, j=8..15 -> K=kb*32+16+(j-8)
//   lane>=16: row M=lane-16, same k-pattern + 8
// -------------------------------------------------------------------------
__global__ void build_weights(const float* __restrict__ omega,
                              const float* __restrict__ alpha_raw,
                              const float* __restrict__ b_log_mag,
                              const float* __restrict__ b_phase,
                              unsigned short* __restrict__ wfrag) {
    int idx  = blockIdx.x * blockDim.x + threadIdx.x;
    int j    = idx & 15;
    int lane = (idx >> 4) & 31;
    int t2   = idx >> 9;
    int kb   = t2 % KB;
    int rb   = t2 / KB;
    if (rb >= RB) return;
    int m      = lane & 15;
    int half   = lane >> 4;
    int klocal = ((j < 8) ? j : j + 8) + half * 8;
    int kk     = kb * 32 + klocal;
    int row    = rb * 16 + m;                   // 0..127 real, 128..255 imag
    int d      = row & (DCH - 1);
    float tau   = (float)(KK - 1 - kk);         // pre-flipped (causal correlation)
    float alpha = -log1pf(expf(alpha_raw[d]));  // -softplus
    float scale = expf(b_log_mag[d] + alpha * tau);
    float phase = b_phase[d] + omega[d] * tau;
    float val   = scale * ((row >= DCH) ? sinf(phase) : cosf(phase));
    wfrag[idx] = f2h_bits(val);
}

// -------------------------------------------------------------------------
// Kernel 2: zero-padded f16 copies of x (even & odd phase) so any 8-element
// window load is 4B-aligned; global b128 in DWORD mode drops only 2 LSBs.
// -------------------------------------------------------------------------
__global__ void build_x(const float* __restrict__ x,
                        unsigned short* __restrict__ xeven,
                        unsigned short* __restrict__ xodd) {
    int b = blockIdx.y;
    int j = blockIdx.x * blockDim.x + threadIdx.x;
    if (j >= SP) return;
    const float* xb = x + (size_t)b * LT;
    int t0 = j - (KK - 1);
    int t1 = t0 + 1;
    float v0 = (t0 >= 0 && t0 < LT) ? xb[t0] : 0.0f;
    float v1 = (t1 >= 0 && t1 < LT) ? xb[t1] : 0.0f;
    xeven[(size_t)b * SP + j] = f2h_bits(v0);
    xodd [(size_t)b * SP + j] = f2h_bits(v1);
}

// -------------------------------------------------------------------------
// Kernel 3: power spectrogram via f16 WMMA GEMM.
// H(256 x NT) = W(256 x 1152) x Xwin(1152 x NT), out = Hr^2 + Hi^2.
// B staged through LDS with 4-deep async pipeline; A software-pipelined
// one kb ahead in ping-pong register sets (2x-unrolled kb loop, no moves).
// -------------------------------------------------------------------------
union AFrag { uint4 q[2]; v16h v; };
union BFrag { uint4 q[2]; v16h v; };

__global__ void __launch_bounds__(256)
sst_power_kernel(const unsigned short* __restrict__ xeven,
                 const unsigned short* __restrict__ xodd,
                 const unsigned short* __restrict__ wfrag,
                 float* __restrict__ out) {
    __shared__ uint4 sbuf[4][256];        // 4 x 4KB B-tiles, async pipeline

    const int tid  = threadIdx.x;
    const int lane = tid & 31;
    const int wave = tid >> 5;            // 8 waves = channel blocks d0 = 16*wave
    const int col  = lane & 15;
    const int half = lane >> 4;

    const int b  = blockIdx.y;
    const int t0 = blockIdx.x * NT;

    // ---- staging role: (s_st, lane_st, chunk c_st) -> one async b128 ----
    const int s_st    = tid >> 6;
    const int lane_st = (tid >> 1) & 31;
    const int c_st    = tid & 1;
    const int off0 = t0 + s_st * 16 + (lane_st & 15) + (lane_st >> 4) * 8 + c_st * 16;
    const unsigned short* xe = xeven + (size_t)b * SP;
    const unsigned short* xo = xodd  + (size_t)b * SP;
    const char* gbase = ((lane_st & 1) ? ((const char*)xo - 2) : (const char*)xe)
                        + 2 * (size_t)off0;                       // 4B-aligned
    const uint32_t ldsbase =
        (uint32_t)(uintptr_t)&sbuf[0][(s_st * 32 + lane_st) * 2 + c_st];

    const unsigned short* wr = wfrag + (size_t)( wave      * KB) * 32 * 16;
    const unsigned short* wi = wfrag + (size_t)((wave + 8) * KB) * 32 * 16;

    v8f cr[4], ci[4];
    const v8f vzero = {0.f,0.f,0.f,0.f,0.f,0.f,0.f,0.f};
    #pragma unroll
    for (int s = 0; s < 4; ++s) { cr[s] = vzero; ci[s] = vzero; }

    auto stage = [&](int kb) {
        uint32_t    ld = ldsbase + (uint32_t)((kb & 3) * 4096);
        const char* gp = gbase + (size_t)kb * 64;                 // 32 elems * 2B
        asm volatile("global_load_async_to_lds_b128 %0, %1, off"
                     :: "v"(ld), "v"(gp) : "memory");
    };
    auto loadA = [&](int kb, AFrag& ar, AFrag& ai) {
        const uint4* pr = (const uint4*)(wr + ((size_t)kb * 32 + lane) * 16);
        ar.q[0] = pr[0]; ar.q[1] = pr[1];
        const uint4* pi = (const uint4*)(wi + ((size_t)kb * 32 + lane) * 16);
        ai.q[0] = pi[0]; ai.q[1] = pi[1];
    };
    auto consume = [&](int kb, const AFrag& ar, const AFrag& ai) {
        const uint4* sb = &sbuf[kb & 3][0];
        BFrag bf[4];                                   // distinct live ranges:
        #pragma unroll                                 // 8 ds_load_b128 up front
        for (int s = 0; s < 4; ++s) {
            bf[s].q[0] = sb[(s * 32 + lane) * 2 + 0];
            bf[s].q[1] = sb[(s * 32 + lane) * 2 + 1];
        }
        #pragma unroll
        for (int s = 0; s < 4; ++s) {
            cr[s] = __builtin_amdgcn_wmma_f32_16x16x32_f16(
                        false, ar.v, false, bf[s].v, (short)0, cr[s], false, false);
            ci[s] = __builtin_amdgcn_wmma_f32_16x16x32_f16(
                        false, ai.v, false, bf[s].v, (short)0, ci[s], false, false);
        }
    };

    AFrag arA, aiA, arB, aiB;
    loadA(0, arA, aiA);                 // A pipeline prologue
    stage(0); stage(1);                 // B async pipeline prologue (depth 2)

    #pragma unroll 1
    for (int kb = 0; kb < KB - 2; kb += 2) {
        // even step: consume kb with set A, prefetch A(kb+1) into set B
        asm volatile("s_wait_asynccnt 0x1" ::: "memory");   // tile kb landed
        __syncthreads();
        stage(kb + 2);
        loadA(kb + 1, arB, aiB);
        consume(kb, arA, aiA);

        // odd step: consume kb+1 with set B, prefetch A(kb+2) into set A
        asm volatile("s_wait_asynccnt 0x1" ::: "memory");   // tile kb+1 landed
        __syncthreads();
        stage(kb + 3);
        loadA(kb + 2, arA, aiA);
        consume(kb + 1, arB, aiB);
    }
    // peeled tail: kb = KB-2, KB-1 (no further staging; exact async counts)
    asm volatile("s_wait_asynccnt 0x1" ::: "memory");
    __syncthreads();
    loadA(KB - 1, arB, aiB);
    consume(KB - 2, arA, aiA);

    asm volatile("s_wait_asynccnt 0x0" ::: "memory");
    __syncthreads();
    consume(KB - 1, arB, aiB);

    // Epilogue: power = Hr^2 + Hi^2; each lane owns 8 consecutive channels.
    float* outb = out + (size_t)b * LT * DCH;
    #pragma unroll
    for (int s = 0; s < 4; ++s) {
        v8f p = cr[s] * cr[s] + ci[s] * ci[s];
        int t = t0 + s * 16 + col;
        float* op = outb + (size_t)t * DCH + wave * 16 + half * 8;  // 32B-aligned
        __builtin_memcpy(op, &p, 32);
    }
}

extern "C" void kernel_launch(void* const* d_in, const int* in_sizes, int n_in,
                              void* d_out, int out_size, void* d_ws, size_t ws_size,
                              hipStream_t stream) {
    const float* x         = (const float*)d_in[0];
    const float* omega     = (const float*)d_in[1];
    const float* alpha_raw = (const float*)d_in[2];
    const float* b_log_mag = (const float*)d_in[3];
    const float* b_phase   = (const float*)d_in[4];
    // d_in[5] = K, fully determined by reference constants (==1152), baked in.
    (void)in_sizes; (void)n_in; (void)out_size; (void)ws_size;

    // workspace: weights (576 KiB) + xeven + xodd (2 x 1.24 MiB) ~= 3.2 MiB
    unsigned short* wfrag = (unsigned short*)d_ws;
    unsigned short* xeven = wfrag + (size_t)RB * KB * 32 * 16;
    unsigned short* xodd  = xeven + (size_t)BB * SP;
    float* out = (float*)d_out;

    build_weights<<<dim3((RB * KB * 32 * 16) / 256), 256, 0, stream>>>(
        omega, alpha_raw, b_log_mag, b_phase, wfrag);
    build_x<<<dim3(SP / 256, BB), 256, 0, stream>>>(x, xeven, xodd);
    sst_power_kernel<<<dim3(LT / NT, BB), 256, 0, stream>>>(xeven, xodd, wfrag, out);
}
